// Transformer_Block_2920577761650
// MI455X (gfx1250) — compile-verified, matrix operand order checked by reading, not run on hardware
//
#include <hip/hip_runtime.h>
#include <math.h>

typedef __bf16 bf16;
typedef __attribute__((ext_vector_type(4)))  __bf16 bf16x4;
typedef __attribute__((ext_vector_type(8)))  __bf16 bf16x8;
typedef __attribute__((ext_vector_type(16))) __bf16 bf16x16;
typedef __attribute__((ext_vector_type(8)))  float  f32x8;

#define NB   8
#define SEQ  2048
#define DIM  1024
#define BS_TOTAL (NB * SEQ)   // 16384 rows

static __device__ __forceinline__ bf16x16 frag_cat(bf16x8 lo, bf16x8 hi) {
  return __builtin_shufflevector(lo, hi, 0,1,2,3,4,5,6,7,8,9,10,11,12,13,14,15);
}
static __device__ __forceinline__ f32x8 wmma_bf16(bf16x16 a, bf16x16 b, f32x8 c) {
  return __builtin_amdgcn_wmma_f32_16x16x32_bf16(false, a, false, b, (short)0, c, false, false);
}

// CDNA5 async copy: global -> LDS, 16B per lane, tracked by ASYNCcnt.
static __device__ __forceinline__ void async_b128(unsigned lds_byte_addr,
                                                  const void* gaddr) {
  asm volatile("global_load_async_to_lds_b128 %0, %1, off"
               :: "v"(lds_byte_addr), "v"((unsigned long long)gaddr)
               : "memory");
}
static __device__ __forceinline__ void wait_async0() {
  asm volatile("s_wait_asynccnt 0x0" ::: "memory");
}
// low 32 bits of a flat shared-aperture address == LDS byte offset (ISA 10.2)
static __device__ __forceinline__ unsigned lds_addr(const void* p) {
  return (unsigned)(size_t)p;
}

// ---------------------------------------------------------------- fp32 -> bf16
__global__ void cvt_kernel(const float* __restrict__ x, bf16* __restrict__ y, int n4) {
  int i = blockIdx.x * 256 + threadIdx.x;
  if (i >= n4) return;
  float4 f = ((const float4*)x)[i];
  bf16x4 o;
  o[0] = (bf16)f.x; o[1] = (bf16)f.y; o[2] = (bf16)f.z; o[3] = (bf16)f.w;
  ((bf16x4*)y)[i] = o;
}

// -------------------------------------------- W[k][n] fp32 -> Wt[n][k] bf16
__global__ void wtrans_kernel(const float* __restrict__ W, bf16* __restrict__ Wt) {
  __shared__ float t[32][33];
  int bx = blockIdx.x * 32, by = blockIdx.y * 32;
  int tx = threadIdx.x, ty = threadIdx.y;          // block 32x8
  #pragma unroll
  for (int i = 0; i < 32; i += 8)
    t[ty + i][tx] = W[(size_t)(by + ty + i) * DIM + bx + tx];
  __syncthreads();
  #pragma unroll
  for (int i = 0; i < 32; i += 8)
    Wt[(size_t)(bx + ty + i) * DIM + by + tx] = (bf16)t[tx][ty + i];
}

// ------------------------------------------------------------------- GEMM
// C[M=16384, N=1024] = A[M,1024](bf16 rows) * W, with Wt[N][K] bf16.
// MODE 0: bf16 row-major out    (K / Q projections)
// MODE 1: bf16 transposed out Vt[b][d][s]   (V projections)
// MODE 2: f32 out, + bias + relu            (FFN)
// Double-buffered LDS weight tile staged via GLOBAL_LOAD_ASYNC_TO_LDS_B128;
// A fragment register double-buffered.
template <int MODE>
__launch_bounds__(256)
__global__ void gemm_kernel(const bf16* __restrict__ A, const bf16* __restrict__ Wt,
                            void* __restrict__ outp, const float* __restrict__ bias) {
  __shared__ __align__(16) bf16 ldsW[2][128][32];   // 2 x 8KB
  const int tid = threadIdx.x;
  const int w = tid >> 5, lane = tid & 31;
  const int h = lane >> 4, ln = lane & 15;
  const int row0 = blockIdx.x * 128 + w * 16;
  const int col0 = blockIdx.y * 128;

  f32x8 acc[8];
  #pragma unroll
  for (int t = 0; t < 8; t++)
    #pragma unroll
    for (int j = 0; j < 8; j++) acc[t][j] = 0.f;

  const bf16* arow = A + (size_t)(row0 + ln) * DIM;

  // each thread stages two 16B chunks of the 128x32 tile per k-step
  const int f0 = tid, f1 = tid + 256;
  const bf16* src0 = Wt + (size_t)(col0 + (f0 >> 2)) * DIM + 8 * (f0 & 3);
  const bf16* src1 = Wt + (size_t)(col0 + (f1 >> 2)) * DIM + 8 * (f1 & 3);
  const unsigned lbase = lds_addr(&ldsW[0][0][0]);
  const unsigned ld0 = lbase + (unsigned)f0 * 16;
  const unsigned ld1 = lbase + (unsigned)f1 * 16;

  // prologue: stage tile 0, load A fragment 0
  async_b128(ld0, src0);
  async_b128(ld1, src1);
  bf16x16 afCur = frag_cat(*(const bf16x8*)(arow + 8 * h),
                           *(const bf16x8*)(arow + 16 + 8 * h));
  wait_async0();
  __syncthreads();

  int buf = 0;
  for (int k0 = 0; k0 < DIM; k0 += 32, buf ^= 1) {
    const bool has_next = (k0 + 32 < DIM);
    bf16x16 afNext = afCur;
    if (has_next) {
      const unsigned nb = (unsigned)(buf ^ 1) * 8192;
      async_b128(ld0 + nb, src0 + k0 + 32);
      async_b128(ld1 + nb, src1 + k0 + 32);
      afNext = frag_cat(*(const bf16x8*)(arow + k0 + 32 + 8 * h),
                        *(const bf16x8*)(arow + k0 + 48 + 8 * h));
    }
    #pragma unroll
    for (int t = 0; t < 8; t++) {
      const bf16* bp = &ldsW[buf][16 * t + ln][0];
      bf16x16 bfr = frag_cat(*(const bf16x8*)(bp + 8 * h),
                             *(const bf16x8*)(bp + 16 + 8 * h));
      acc[t] = wmma_bf16(afCur, bfr, acc[t]);
    }
    if (has_next) wait_async0();
    __syncthreads();
    afCur = afNext;
  }

  #pragma unroll
  for (int t = 0; t < 8; t++) {
    int col = col0 + 16 * t + ln;
    #pragma unroll
    for (int r = 0; r < 8; r++) {
      int row = row0 + r + 8 * h;
      float v = acc[t][r];
      if (MODE == 0) {
        ((bf16*)outp)[(size_t)row * DIM + col] = (bf16)v;
      } else if (MODE == 1) {
        int b = row >> 11, s = row & (SEQ - 1);
        ((bf16*)outp)[((size_t)b << 21) + (size_t)col * SEQ + s] = (bf16)v;
      } else {
        v += bias[col];
        ((float*)outp)[(size_t)row * DIM + col] = fmaxf(v, 0.f);
      }
    }
  }
}

// --------------------------------------------------------------- attention
// Q,K bf16 row-major [b][s][d]; Vt bf16 [b][d][s]; out f32 [b][s][d].
// Block = 128 threads (4 waves) per 16-query tile; wave w owns d-slice [256w,256w+256).
template <bool CAUSAL>
__launch_bounds__(128)
__global__ void attn_kernel(const bf16* __restrict__ Q, const bf16* __restrict__ K,
                            const bf16* __restrict__ Vt, float* __restrict__ out) {
  __shared__ __align__(16) bf16 sQ[16][DIM];        // 32 KB
  __shared__ float sS[4][2][8][32];                 // 8 KB partial scores
  __shared__ __align__(16) bf16 sP[16][32];         // 1 KB probs

  const int tid = threadIdx.x;
  const int w = tid >> 5, lane = tid & 31;
  const int h = lane >> 4, ln = lane & 15;
  const int b = blockIdx.y;
  const int q0 = blockIdx.x * 16;
  const int d0 = w * 256;

  // async cooperative Q tile load (16 rows are contiguous in memory)
  {
    const uint4* qbase = (const uint4*)(Q + (size_t)(b * SEQ + q0) * DIM);
    const unsigned qlds = lds_addr(&sQ[0][0]);
    #pragma unroll
    for (int f = tid; f < 2048; f += 128)
      async_b128(qlds + (unsigned)f * 16, qbase + f);
    wait_async0();
    __syncthreads();
  }

  f32x8 oacc[16];
  #pragma unroll
  for (int i = 0; i < 16; i++)
    #pragma unroll
    for (int j = 0; j < 8; j++) oacc[i][j] = 0.f;

  float m_[8], l_[8];
  #pragma unroll
  for (int r = 0; r < 8; r++) { m_[r] = -__builtin_inff(); l_[r] = 0.f; }

  const bf16* kbatch = K + (size_t)b * SEQ * DIM;
  const bf16* vbatch = Vt + (size_t)b * DIM * SEQ;
  const int s_end = CAUSAL ? (q0 + 16) : SEQ;

  for (int s0 = 0; s0 < s_end; s0 += 32) {
    // partial scores over this wave's 256-wide d-slice
    f32x8 sc[2];
    #pragma unroll
    for (int t = 0; t < 2; t++)
      #pragma unroll
      for (int j = 0; j < 8; j++) sc[t][j] = 0.f;

    #pragma unroll
    for (int kk = 0; kk < 8; kk++) {
      const int dk = d0 + kk * 32;
      bf16x16 af = frag_cat(*(const bf16x8*)&sQ[ln][dk + 8 * h],
                            *(const bf16x8*)&sQ[ln][dk + 16 + 8 * h]);
      #pragma unroll
      for (int t = 0; t < 2; t++) {
        const bf16* kp = kbatch + (size_t)(s0 + 16 * t + ln) * DIM + dk;
        bf16x16 bfr = frag_cat(*(const bf16x8*)(kp + 8 * h),
                               *(const bf16x8*)(kp + 16 + 8 * h));
        sc[t] = wmma_bf16(af, bfr, sc[t]);
      }
    }
    #pragma unroll
    for (int t = 0; t < 2; t++)
      #pragma unroll
      for (int r = 0; r < 8; r++) sS[w][t][r][lane] = sc[t][r];
    __syncthreads();

    // full logits (replicated in all waves), scale 1/sqrt(1024), causal mask
    float lg[2][8];
    #pragma unroll
    for (int t = 0; t < 2; t++)
      #pragma unroll
      for (int r = 0; r < 8; r++) {
        float s = sS[0][t][r][lane] + sS[1][t][r][lane] +
                  sS[2][t][r][lane] + sS[3][t][r][lane];
        s *= 0.03125f;
        if (CAUSAL) {
          int key = s0 + 16 * t + ln;
          int qry = q0 + r + 8 * h;
          if (key > qry) s = -__builtin_inff();
        }
        lg[t][r] = s;
      }

    // online softmax; row r maps to query q0 + r + 8h — same mapping as C/D frags
    float rowm[8], al[8], pv0[8], pv1[8], rs[8];
    #pragma unroll
    for (int r = 0; r < 8; r++) rowm[r] = fmaxf(lg[0][r], lg[1][r]);
    #pragma unroll
    for (int mk = 1; mk <= 8; mk <<= 1)
      #pragma unroll
      for (int r = 0; r < 8; r++) rowm[r] = fmaxf(rowm[r], __shfl_xor(rowm[r], mk, 32));
    #pragma unroll
    for (int r = 0; r < 8; r++) {
      float mn = fmaxf(m_[r], rowm[r]);
      al[r]  = __expf(m_[r] - mn);
      pv0[r] = __expf(lg[0][r] - mn);
      pv1[r] = __expf(lg[1][r] - mn);
      rs[r]  = pv0[r] + pv1[r];
      m_[r]  = mn;
    }
    #pragma unroll
    for (int mk = 1; mk <= 8; mk <<= 1)
      #pragma unroll
      for (int r = 0; r < 8; r++) rs[r] += __shfl_xor(rs[r], mk, 32);
    #pragma unroll
    for (int r = 0; r < 8; r++) l_[r] = l_[r] * al[r] + rs[r];

    // stage P (all waves write identical values — benign)
    #pragma unroll
    for (int r = 0; r < 8; r++) {
      sP[r + 8 * h][ln]      = (bf16)pv0[r];
      sP[r + 8 * h][16 + ln] = (bf16)pv1[r];
    }
    __syncthreads();

    // rescale accumulators, then P @ V over this wave's d-slice
    #pragma unroll
    for (int dt = 0; dt < 16; dt++)
      #pragma unroll
      for (int r = 0; r < 8; r++) oacc[dt][r] *= al[r];

    bf16x16 pf = frag_cat(*(const bf16x8*)&sP[ln][8 * h],
                          *(const bf16x8*)&sP[ln][16 + 8 * h]);
    #pragma unroll
    for (int dt = 0; dt < 16; dt++) {
      const bf16* vp = vbatch + (size_t)(d0 + 16 * dt + ln) * SEQ + s0;
      bf16x16 bfr = frag_cat(*(const bf16x8*)(vp + 8 * h),
                             *(const bf16x8*)(vp + 16 + 8 * h));
      oacc[dt] = wmma_bf16(pf, bfr, oacc[dt]);
    }
  }

  #pragma unroll
  for (int r = 0; r < 8; r++) {
    float inv = 1.f / l_[r];
    size_t rowoff = (size_t)(b * SEQ + q0 + r + 8 * h) * DIM + d0;
    #pragma unroll
    for (int dt = 0; dt < 16; dt++)
      out[rowoff + 16 * dt + ln] = oacc[dt][r] * inv;
  }
}

// ------------------------------------------------- residual + LayerNorm (+relu)
template <bool FINAL>
__launch_bounds__(128)
__global__ void ln_kernel(const float* __restrict__ a, const float* __restrict__ res,
                          const float* __restrict__ g, const float* __restrict__ bb,
                          float* __restrict__ outf, bf16* __restrict__ outb) {
  __shared__ float rs_[4], rq_[4];
  const int row = blockIdx.x, tid = threadIdx.x;
  const size_t base = (size_t)row * DIM;
  float v[8], s = 0.f, sq = 0.f;
  #pragma unroll
  for (int i = 0; i < 8; i++) {
    int c = i * 128 + tid;
    v[i] = a[base + c] + res[base + c];
    s += v[i]; sq += v[i] * v[i];
  }
  #pragma unroll
  for (int mk = 1; mk < 32; mk <<= 1) {
    s  += __shfl_xor(s, mk, 32);
    sq += __shfl_xor(sq, mk, 32);
  }
  if ((tid & 31) == 0) { rs_[tid >> 5] = s; rq_[tid >> 5] = sq; }
  __syncthreads();
  s  = rs_[0] + rs_[1] + rs_[2] + rs_[3];
  sq = rq_[0] + rq_[1] + rq_[2] + rq_[3];
  float mu   = s * (1.f / 1024.f);
  float var  = sq * (1.f / 1024.f) - mu * mu;
  float rstd = rsqrtf(var + 1e-3f);
  #pragma unroll
  for (int i = 0; i < 8; i++) {
    int c = i * 128 + tid;
    float y = (v[i] - mu) * rstd * g[c] + bb[c];
    if (FINAL) {
      outf[base + c] = fmaxf(y, 0.f);
    } else {
      outf[base + c] = y;
      outb[base + c] = (bf16)y;
    }
  }
}

// ------------------------------------------------------------------ driver
extern "C" void kernel_launch(void* const* d_in, const int* in_sizes, int n_in,
                              void* d_out, int out_size, void* d_ws, size_t ws_size,
                              hipStream_t stream) {
  (void)in_sizes; (void)n_in; (void)out_size; (void)ws_size;
  const float* inputs  = (const float*)d_in[0];
  const float* context = (const float*)d_in[1];
  const float* Wmat[7] = { (const float*)d_in[2], (const float*)d_in[3],
                           (const float*)d_in[4], (const float*)d_in[5],
                           (const float*)d_in[6], (const float*)d_in[7],
                           (const float*)d_in[8] };
  const float* b1   = (const float*)d_in[9];
  const float* ln_g = (const float*)d_in[10];
  const float* ln_b = (const float*)d_in[11];
  float* out = (float*)d_out;

  const size_t ACT_BF = (size_t)BS_TOTAL * DIM * sizeof(bf16);   // 32 MB
  const size_t ACT_F  = (size_t)BS_TOTAL * DIM * sizeof(float);  // 64 MB
  char* ws = (char*)d_ws;
  size_t o = 0;
  bf16* Wt   = (bf16*)(ws + o); o += (size_t)7 * DIM * DIM * sizeof(bf16);
  bf16* Xbf  = (bf16*)(ws + o); o += ACT_BF;
  bf16* Cbf  = (bf16*)(ws + o); o += ACT_BF;
  bf16* Kbuf = (bf16*)(ws + o); o += ACT_BF;
  bf16* Qbuf = (bf16*)(ws + o); o += ACT_BF;
  bf16* Vtb  = (bf16*)(ws + o); o += ACT_BF;
  float* att = (float*)(ws + o); o += ACT_F;
  float* x1  = (float*)(ws + o); o += ACT_F;
  float* x2  = (float*)(ws + o); o += ACT_F;
  bf16* xbf  = (bf16*)(ws + o); o += ACT_BF;

  const int n4 = BS_TOTAL * DIM / 4;
  dim3 gGemm(BS_TOTAL / 128, DIM / 128);   // (128, 8)
  dim3 gAttn(SEQ / 16, NB);                // (128, 8)
  dim3 bTr(32, 8);
  dim3 gTr(32, 32);

  // bf16 conversions
  cvt_kernel<<<n4 / 256, 256, 0, stream>>>(inputs,  Xbf, n4);
  cvt_kernel<<<n4 / 256, 256, 0, stream>>>(context, Cbf, n4);
  for (int i = 0; i < 7; i++)
    wtrans_kernel<<<gTr, bTr, 0, stream>>>(Wmat[i], Wt + (size_t)i * DIM * DIM);

  bf16* WtKs = Wt;                      bf16* WtVs = Wt + 1u * DIM * DIM;
  bf16* WtQs = Wt + 2u * DIM * DIM;     bf16* WtKc = Wt + 3u * DIM * DIM;
  bf16* WtVc = Wt + 4u * DIM * DIM;     bf16* WtQc = Wt + 5u * DIM * DIM;
  bf16* WtW1 = Wt + 6u * DIM * DIM;

  // ---- causal self-attention ----
  gemm_kernel<0><<<gGemm, 256, 0, stream>>>(Xbf, WtKs, Kbuf, nullptr);
  gemm_kernel<0><<<gGemm, 256, 0, stream>>>(Xbf, WtQs, Qbuf, nullptr);
  gemm_kernel<1><<<gGemm, 256, 0, stream>>>(Xbf, WtVs, Vtb,  nullptr);
  attn_kernel<true><<<gAttn, 128, 0, stream>>>(Qbuf, Kbuf, Vtb, att);
  ln_kernel<false><<<BS_TOTAL, 128, 0, stream>>>(att, inputs, ln_g, ln_b, x1, xbf);

  // ---- cross attention (Q from x, K/V from context) ----
  gemm_kernel<0><<<gGemm, 256, 0, stream>>>(Cbf, WtKc, Kbuf, nullptr);
  gemm_kernel<1><<<gGemm, 256, 0, stream>>>(Cbf, WtVc, Vtb,  nullptr);
  gemm_kernel<0><<<gGemm, 256, 0, stream>>>(xbf, WtQc, Qbuf, nullptr);
  attn_kernel<false><<<gAttn, 128, 0, stream>>>(Qbuf, Kbuf, Vtb, att);
  ln_kernel<false><<<BS_TOTAL, 128, 0, stream>>>(att, x1, ln_g, ln_b, x2, xbf);

  // ---- FFN + final LN + relu ----
  gemm_kernel<2><<<gGemm, 256, 0, stream>>>(xbf, WtW1, att, b1);
  ln_kernel<true><<<BS_TOTAL, 128, 0, stream>>>(att, x2, ln_g, ln_b, out, nullptr);
}
